// RNN_Dale_83476984365507
// MI455X (gfx1250) — matrix operand order repaired and checked by reading, not compiled, not run on hardware
//
#include <hip/hip_runtime.h>
#include <math.h>

// ---------------------------------------------------------------------------
// RNN with Dale's law, persistent-kernel formulation for MI455X (gfx1250).
//
// 2048 sequentially-dependent steps of a 128x512x512 GEMM. Latency-bound
// chain => persistent kernel (8 WGs, one per WGP), Wrec^T slice pinned in
// LDS in f16 for all steps, v_wmma_f32_16x16x32_f16 for the GEMM, pre-relu
// f32 state held entirely in registers (static tile ownership), relu'd f16
// activity published via 2-byte stores and re-staged with
// GLOBAL_LOAD_ASYNC_TO_LDS_B128 (ASYNCcnt) each step. One device-wide
// barrier per step.
// ---------------------------------------------------------------------------

#define NU    512          // units
#define BB    128          // batch
#define TT    2048         // timesteps
#define NEXC  409          // int(0.8 * 512) excitatory units
#define NWG   8            // persistent workgroups (one per WGP)
#define NSL   (NU / NWG)   // 64 output units per WG
#define LSTR  520          // padded LDS row stride in f16 (512+8 -> 4-bank rotate/row)
#define POLD  0.8f
#define PNEW  0.2f
#define NCONST 0.0632455532033676f   // sqrt(2*0.01/5)

typedef __attribute__((ext_vector_type(16))) _Float16 v16h;
typedef __attribute__((ext_vector_type(8)))  _Float16 v8h;
typedef __attribute__((ext_vector_type(4)))  _Float16 v4h;
typedef __attribute__((ext_vector_type(8)))  float    v8f;
typedef __attribute__((ext_vector_type(4)))  float    v4f;

__global__ void rnn_init_kernel(unsigned* bar) {
    bar[0] = 0u;   // arrival counter
    bar[1] = 0u;   // generation
}

__device__ __forceinline__ void grid_barrier(unsigned* cnt, unsigned* gen) {
    __threadfence();            // release: make this WG's global stores visible
    __syncthreads();
    if (threadIdx.x == 0) {
        unsigned g = __hip_atomic_load(gen, __ATOMIC_RELAXED, __HIP_MEMORY_SCOPE_AGENT);
        unsigned old = __hip_atomic_fetch_add(cnt, 1u, __ATOMIC_ACQ_REL, __HIP_MEMORY_SCOPE_AGENT);
        if (old == NWG - 1) {
            __hip_atomic_store(cnt, 0u, __ATOMIC_RELAXED, __HIP_MEMORY_SCOPE_AGENT);
            __hip_atomic_fetch_add(gen, 1u, __ATOMIC_RELEASE, __HIP_MEMORY_SCOPE_AGENT);
        } else {
            while (__hip_atomic_load(gen, __ATOMIC_ACQUIRE, __HIP_MEMORY_SCOPE_AGENT) == g) {
                __builtin_amdgcn_s_sleep(1);
            }
        }
    }
    __syncthreads();            // acquire for all threads in WG
}

__global__ __launch_bounds__(256)
void rnn_dale_persistent(const float* __restrict__ inputs,   // [B,1,T]
                         const float* __restrict__ noise,    // [T,B,N]
                         const float* __restrict__ inp,      // [1,N]
                         const float* __restrict__ wrecdale, // [N,N]
                         const float* __restrict__ out_w,    // [N,1]
                         _Float16* __restrict__ actH0,       // [B,N] relu(act) f16 ping
                         _Float16* __restrict__ actH1,       // [B,N] relu(act) f16 pong
                         unsigned* __restrict__ bar,
                         float* __restrict__ out)            // [B] ++ [B,N]
{
    // LDS: full relu(activity) operand (f16) + this WG's Wrec^T slice (f16).
    __shared__ _Float16 sA[BB  * LSTR];   // [b][k]        ~130 KB
    __shared__ _Float16 sW[NSL * LSTR];   // [n_local][k]  ~65 KB   (B matrix, [k][n] logical)

    const int tid  = threadIdx.x;
    const int lane = tid & 31;
    const int wave = tid >> 5;            // 8 waves = 8 M-tiles of 16 batch rows
    const int g    = lane >> 4;           // half-wave (0/1)
    const int c    = lane & 15;
    const int wg   = blockIdx.x;
    const int n0   = wg * NSL;            // this WG's output-unit slice
    const int mtile = wave;

    // ---- one-time: load Wrec^T slice into LDS: sW[nl][k] = |Wd[n0+nl][k]| * sdale[k]
    #pragma unroll 4
    for (int i = 0; i < (NSL * NU) / (256 * 4); ++i) {   // 32 iters, 4 floats each
        int f  = i * 1024 + tid * 4;
        int nl = f >> 9;
        int k  = f & (NU - 1);
        v4f w  = *(const v4f*)(wrecdale + (size_t)(n0 + nl) * NU + k);
        v4h h;
        #pragma unroll
        for (int j = 0; j < 4; ++j) {
            float s = ((k + j) < NEXC) ? 1.0f : -1.0f;
            h[j] = (_Float16)(fabsf(w[j]) * s);
        }
        *(v4h*)(&sW[nl * LSTR + k]) = h;
    }

    // ---- per-thread register state: pre-relu activity for owned (b,n) tiles.
    //   element (nt, r) -> b = mtile*16 + 8*g + r ; n = n0 + nt*16 + c
    float st[4][8];
    #pragma unroll
    for (int nt = 0; nt < 4; ++nt)
        #pragma unroll
        for (int r = 0; r < 8; ++r) st[nt][r] = 0.0f;

    // ---- one-time: publish relu(act0)=0 (f16) for step 0's staging
    #pragma unroll
    for (int nt = 0; nt < 4; ++nt) {
        const int n = n0 + nt * 16 + c;
        #pragma unroll
        for (int r = 0; r < 8; ++r) {
            const int b = mtile * 16 + g * 8 + r;
            actH0[b * NU + n] = (_Float16)0.0f;
        }
    }
    grid_barrier(bar, bar + 1);

    // ---- recurrent scan
    #pragma unroll 1
    for (int t = 0; t < TT; ++t) {
        const _Float16* hOld = (t & 1) ? actH1 : actH0;
        _Float16*       hNew = (t & 1) ? actH0 : actH1;

        // stage relu(activity) f16 [128][512] into padded LDS via the CDNA5
        // async-to-LDS path (no VGPR round trip, tracked by ASYNCcnt).
        #pragma unroll 4
        for (int i = 0; i < (BB * NU) / (256 * 8); ++i) {  // 32 iters, 16 B each
            int f   = i * 2048 + tid * 8;
            int row = f >> 9;
            int col = f & (NU - 1);
            unsigned ldsoff = (unsigned)(size_t)(&sA[row * LSTR + col]);
            const _Float16* gp = hOld + f;
            asm volatile("global_load_async_to_lds_b128 %0, %1, off"
                         :: "v"(ldsoff), "v"(gp) : "memory");
        }
        // prefetch next step's noise rows for this WG's unit slice
        if (t + 1 < TT)
            __builtin_prefetch(noise + ((size_t)(t + 1) * BB + (tid & (BB - 1))) * NU + n0, 0, 0);
        asm volatile("s_wait_asynccnt 0x0" ::: "memory");
        __syncthreads();

        // ---- WMMA: rec[b, n] = sum_k relu(a[b,k]) * Wrec[n,k]
        v8f acc[4];
        #pragma unroll
        for (int nt = 0; nt < 4; ++nt) acc[nt] = (v8f)(0.0f);

        #pragma unroll 4
        for (int kk = 0; kk < NU / 32; ++kk) {        // 16 K-steps of 32
            const int k0 = kk * 32;
            // A fragment (16x32 f16): lane holds row M=c, K in two 8-wide runs
            const _Float16* ap = &sA[(mtile * 16 + c) * LSTR + k0 + g * 8];
            v8h alo = *(const v8h*)(ap);
            v8h ahi = *(const v8h*)(ap + 16);
            v16h a  = __builtin_shufflevector(alo, ahi,
                        0,1,2,3,4,5,6,7,8,9,10,11,12,13,14,15);
            #pragma unroll
            for (int nt = 0; nt < 4; ++nt) {
                // B fragment (32x16 f16): lane holds col N=c, K in two 8-wide runs
                const _Float16* bp = &sW[(nt * 16 + c) * LSTR + k0 + g * 8];
                v8h blo = *(const v8h*)(bp);
                v8h bhi = *(const v8h*)(bp + 16);
                v16h b  = __builtin_shufflevector(blo, bhi,
                            0,1,2,3,4,5,6,7,8,9,10,11,12,13,14,15);
                acc[nt] = __builtin_amdgcn_wmma_f32_16x16x32_f16(
                            false, a, false, b, (short)0, acc[nt], false, false);
            }
        }

        // ---- fused state update in the C-matrix layout (all in registers):
        float uval[8];
        #pragma unroll
        for (int r = 0; r < 8; ++r)
            uval[r] = inputs[(size_t)(mtile * 16 + g * 8 + r) * TT + t];

        #pragma unroll
        for (int nt = 0; nt < 4; ++nt) {
            const int n    = n0 + nt * 16 + c;
            const float ia = fabsf(inp[n]);
            #pragma unroll
            for (int r = 0; r < 8; ++r) {
                const int b = mtile * 16 + g * 8 + r;
                float nz = noise[((size_t)t * BB + b) * NU + n];
                float v  = POLD * st[nt][r] + PNEW * (acc[nt][r] + uval[r] * ia)
                         + NCONST * nz;
                st[nt][r] = v;
                hNew[b * NU + n] = (_Float16)fmaxf(v, 0.0f);   // publish relu f16
            }
        }
        grid_barrier(bar, bar + 1);    // release stores, sync step t
    }

    // ---- emit final activity (f32, straight from registers) into d_out
    #pragma unroll
    for (int nt = 0; nt < 4; ++nt) {
        const int n = n0 + nt * 16 + c;
        #pragma unroll
        for (int r = 0; r < 8; ++r) {
            const int b = mtile * 16 + g * 8 + r;
            out[BB + b * NU + n] = st[nt][r];
        }
    }
    grid_barrier(bar, bar + 1);

    // ---- readout: output[b] = act_final[b,:] . out_w
    if (wg == 0) {
        const float* actF = out + BB;
        for (int b = tid; b < BB; b += 256) {
            float s = 0.0f;
            for (int n = 0; n < NU; ++n) s += actF[b * NU + n] * out_w[n];
            out[b] = s;
        }
    }
}

extern "C" void kernel_launch(void* const* d_in, const int* in_sizes, int n_in,
                              void* d_out, int out_size, void* d_ws, size_t ws_size,
                              hipStream_t stream) {
    const float* inputs   = (const float*)d_in[0];   // [128,1,2048]
    const float* noise    = (const float*)d_in[1];   // [2048,128,512]
    const float* inp      = (const float*)d_in[2];   // [1,512]
    const float* wrecdale = (const float*)d_in[3];   // [512,512]
    const float* out_w    = (const float*)d_in[4];   // [512,1]
    float* out = (float*)d_out;                      // [128] ++ [128,512]

    _Float16* actH0 = (_Float16*)d_ws;               // 128 KB relu(act) ping
    _Float16* actH1 = actH0 + BB * NU;               // 128 KB relu(act) pong
    unsigned* bar   = (unsigned*)(actH1 + BB * NU);  // 8 B barrier state

    rnn_init_kernel<<<1, 1, 0, stream>>>(bar);
    rnn_dale_persistent<<<NWG, 256, 0, stream>>>(inputs, noise, inp, wrecdale,
                                                 out_w, actH0, actH1, bar, out);
}
